// Earth4DModel_32220844654677
// MI455X (gfx1250) — matrix operand order repaired and checked by reading, not compiled, not run on hardware
//
#include <hip/hip_runtime.h>
#include <hip/hip_bf16.h>

typedef __attribute__((ext_vector_type(16))) _Float16 v16h;
typedef __attribute__((ext_vector_type(8)))  float    v8f;

#define SL 36
#define TL 20
#define FEAT 2
#define HID 128
#define ENC_DIM ((SL + TL) * FEAT)   // 112
#define KPAD 128
#define TILE_M 128
#define NTHREADS 256
#define KSTEPS 4                      // KPAD / 32
#define NTILES 8                      // HID / 16
#define ST_MASK ((1u << 19) - 1u)
#define TT_MASK ((1u << 17) - 1u)

union U32H2 { unsigned int u; _Float16 h[2]; };

// ---------------------------------------------------------------------------
// Pack an f32 [K x 128] weight matrix into f16 B-fragments laid out so that
// lane L of a wave can load its v16h operand for v_wmma_f32_16x16x32_f16 with
// two contiguous 128-bit loads.  B packing used here (32x16, k contiguous per lane half):
//   col = ntile*16 + (lane & 15);  k = kstep*32 + (lane>>4)*16 + h,  h=0..15
// dst index = (((ntile*KSTEPS + kstep)*32 + lane)*16 + h)
// Rows k >= K are zero padding (K=112 for w1).
// ---------------------------------------------------------------------------
__global__ void pack_w_f16(const float* __restrict__ w, _Float16* __restrict__ dst,
                           int K) {
    int idx = blockIdx.x * blockDim.x + threadIdx.x;
    const int total = NTILES * KSTEPS * 32 * 16;
    if (idx >= total) return;
    int h     = idx & 15;
    int lane  = (idx >> 4) & 31;
    int step  = (idx >> 9) & (KSTEPS - 1);
    int ntile = (idx >> 9) >> 2;
    int col = ntile * 16 + (lane & 15);
    int k   = step * 32 + ((lane >> 4) << 4) + h;
    float v = (k < K) ? w[k * HID + col] : 0.0f;
    dst[idx] = (_Float16)v;
}

// ---------------------------------------------------------------------------
// Fused: hash-grid encode -> GEMM1(WMMA f16->f32) -> relu -> LN ->
//        GEMM2(WMMA) -> relu -> LN -> dot(w3) + b3
// One block = 128 points, 8 wave32.
// ---------------------------------------------------------------------------
__global__ __launch_bounds__(NTHREADS, 2)
void earth4d_fused(const float* __restrict__ coords,
                   const float* __restrict__ spatial_table,   // [SL][2^19][2]
                   const float* __restrict__ temporal_table,  // [TL][2^17][2]
                   const float* __restrict__ b1,
                   const float* __restrict__ g1,
                   const float* __restrict__ beta1,
                   const float* __restrict__ b2,
                   const float* __restrict__ g2,
                   const float* __restrict__ beta2,
                   const float* __restrict__ w3,
                   const float* __restrict__ b3,
                   const _Float16* __restrict__ pw1,          // packed fragments
                   const _Float16* __restrict__ pw2,
                   float* __restrict__ out) {
    __shared__ _Float16 featA[TILE_M * KPAD];   // 32 KB: A operand (reused per layer)
    __shared__ float    hbuf [TILE_M * HID];    // 64 KB: pre-layernorm activations

    const int tid  = threadIdx.x;
    const int lane = tid & 31;
    const int wave = tid >> 5;
    const int half = lane >> 4;
    const int base = blockIdx.x * TILE_M;

    // ---------------- Stage 0: zero the K padding (cols 112..127) -----------
    for (int t = tid; t < TILE_M * (KPAD - ENC_DIM); t += NTHREADS) {
        int row = t / (KPAD - ENC_DIM);
        int j   = t % (KPAD - ENC_DIM);
        featA[row * KPAD + ENC_DIM + j] = (_Float16)0.0f;
    }

    // ---------------- Stage 1a: spatial hash encoding ----------------------
    const float LOG_SG = 0.32208350f;           // ln(1.38)
    for (int t = tid; t < TILE_M * SL; t += NTHREADS) {
        int row   = t & (TILE_M - 1);
        int level = t >> 7;                     // t / 128
        const float* c = coords + (size_t)(base + row) * 4;
        float res = floorf(16.0f * __expf(LOG_SG * (float)level));
        float fx = c[0] * res, fy = c[1] * res, fz = c[2] * res;
        float x0f = floorf(fx), y0f = floorf(fy), z0f = floorf(fz);
        float wx = fx - x0f, wy = fy - y0f, wz = fz - z0f;
        unsigned int x0 = (unsigned int)(int)x0f;
        unsigned int y0 = (unsigned int)(int)y0f;
        unsigned int z0 = (unsigned int)(int)z0f;
        const float* tab = spatial_table + ((size_t)level << 19) * FEAT;
        float a0 = 0.0f, a1 = 0.0f;
        #pragma unroll
        for (int cr = 0; cr < 8; ++cr) {
            unsigned int dx = cr & 1u, dy = (cr >> 1) & 1u, dz = (cr >> 2) & 1u;
            unsigned int h = ((x0 + dx) * 1u)
                           ^ ((y0 + dy) * 2654435761u)
                           ^ ((z0 + dz) * 805459861u);
            h &= ST_MASK;
            const float* f = tab + (size_t)h * FEAT;
            float wt = (dx ? wx : 1.0f - wx) * (dy ? wy : 1.0f - wy)
                     * (dz ? wz : 1.0f - wz);
            a0 = fmaf(f[0], wt, a0);
            a1 = fmaf(f[1], wt, a1);
        }
        featA[row * KPAD + level * 2 + 0] = (_Float16)a0;
        featA[row * KPAD + level * 2 + 1] = (_Float16)a1;
    }

    // ---------------- Stage 1b: temporal encoding --------------------------
    const float LOG_TG = 0.40546511f;           // ln(1.5)
    for (int t = tid; t < TILE_M * TL; t += NTHREADS) {
        int row   = t & (TILE_M - 1);
        int level = t >> 7;
        float tc  = coords[(size_t)(base + row) * 4 + 3];
        float res = floorf(16.0f * __expf(LOG_TG * (float)level));
        float tx  = tc * res;
        float t0f = floorf(tx);
        float tw  = tx - t0f;
        unsigned int t0 = (unsigned int)(int)t0f;
        unsigned int i0 = t0 & TT_MASK;
        unsigned int i1 = (t0 + 1u) & TT_MASK;
        const float* tab = temporal_table + ((size_t)level << 17) * FEAT;
        const float* f0 = tab + (size_t)i0 * FEAT;
        const float* f1 = tab + (size_t)i1 * FEAT;
        featA[row * KPAD + SL * FEAT + level * 2 + 0] =
            (_Float16)(f0[0] * (1.0f - tw) + f1[0] * tw);
        featA[row * KPAD + SL * FEAT + level * 2 + 1] =
            (_Float16)(f0[1] * (1.0f - tw) + f1[1] * tw);
    }
    __syncthreads();

    // ---------------- GEMM + relu + layernorm, two layers ------------------
    const int m0   = wave * 16;
    const int arow = m0 + (lane & 15);          // A-fragment source row
    const int ncol = lane & 15;                 // C/D column within tile

    #pragma unroll 1
    for (int layer = 0; layer < 2; ++layer) {
        const _Float16* pw   = (layer == 0) ? pw1   : pw2;
        const float*    bb   = (layer == 0) ? b1    : b2;
        const float*    gg   = (layer == 0) ? g1    : g2;
        const float*    bet  = (layer == 0) ? beta1 : beta2;

        const unsigned int* arowp =
            (const unsigned int*)(featA + arow * KPAD);

        #pragma unroll 1
        for (int nt = 0; nt < NTILES; ++nt) {
            const int col = nt * 16 + ncol;
            float bias = bb[col];
            v8f cacc;
            #pragma unroll
            for (int r = 0; r < 8; ++r) cacc[r] = bias;

            #pragma unroll
            for (int ks = 0; ks < KSTEPS; ++ks) {
                // A fragment: 16x32 f16, k = 16*half + element (contiguous, same as packed B)
                v16h afrag;
                #pragma unroll
                for (int v = 0; v < 8; ++v) {
                    int kb = ks * 32 + (half << 4) + (v << 1);
                    U32H2 cv; cv.u = arowp[kb >> 1];
                    afrag[2 * v + 0] = cv.h[0];
                    afrag[2 * v + 1] = cv.h[1];
                }
                // B fragment: contiguous pre-packed v16h (2 x b128 loads)
                v16h bfrag = *(const v16h*)(pw + ((size_t)(nt * KSTEPS + ks) * 32 + lane) * 16);

                cacc = __builtin_amdgcn_wmma_f32_16x16x32_f16(
                    false, afrag, false, bfrag, (short)0, cacc, false, false);
            }
            // relu + stash pre-LN activations (C/D layout: M = half*8 + r)
            #pragma unroll
            for (int r = 0; r < 8; ++r) {
                int row = m0 + half * 8 + r;
                hbuf[row * HID + col] = fmaxf(cacc[r], 0.0f);
            }
        }
        __syncthreads();

        // LayerNorm per row (population variance, eps 1e-5), write f16 A
        if (tid < TILE_M) {
            const float* hr = hbuf + tid * HID;
            float m = 0.0f, s = 0.0f;
            #pragma unroll 4
            for (int k = 0; k < HID; ++k) { float v = hr[k]; m += v; s += v * v; }
            m *= (1.0f / HID);
            float var  = s * (1.0f / HID) - m * m;
            float rinv = rsqrtf(var + 1e-5f);
            #pragma unroll 4
            for (int k = 0; k < HID; ++k)
                featA[tid * KPAD + k] = (_Float16)((hr[k] - m) * rinv * gg[k] + bet[k]);
        }
        __syncthreads();
    }

    // ---------------- Layer 3: [128] . w3 + b3 -----------------------------
    if (tid < TILE_M) {
        float acc = b3[0];
        const _Float16* hr = featA + tid * KPAD;
        #pragma unroll 4
        for (int k = 0; k < HID; ++k) acc = fmaf((float)hr[k], w3[k], acc);
        out[base + tid] = acc;
    }
}

extern "C" void kernel_launch(void* const* d_in, const int* in_sizes, int n_in,
                              void* d_out, int out_size, void* d_ws, size_t ws_size,
                              hipStream_t stream) {
    (void)n_in; (void)out_size; (void)ws_size;
    const float* coords   = (const float*)d_in[0];
    const float* sp_tab   = (const float*)d_in[1];
    const float* tm_tab   = (const float*)d_in[2];
    const float* w1       = (const float*)d_in[3];
    const float* b1       = (const float*)d_in[4];
    const float* g1       = (const float*)d_in[5];
    const float* beta1    = (const float*)d_in[6];
    const float* w2       = (const float*)d_in[7];
    const float* b2       = (const float*)d_in[8];
    const float* g2       = (const float*)d_in[9];
    const float* beta2    = (const float*)d_in[10];
    const float* w3       = (const float*)d_in[11];
    const float* b3       = (const float*)d_in[12];
    float* out = (float*)d_out;

    const int n = in_sizes[0] / 4;              // number of points (131072)

    _Float16* pw1 = (_Float16*)d_ws;
    _Float16* pw2 = pw1 + NTILES * KSTEPS * 32 * 16;

    const int packN = NTILES * KSTEPS * 32 * 16;   // 16384 fragments elements
    pack_w_f16<<<(packN + 255) / 256, 256, 0, stream>>>(w1, pw1, ENC_DIM);
    pack_w_f16<<<(packN + 255) / 256, 256, 0, stream>>>(w2, pw2, HID);

    earth4d_fused<<<n / TILE_M, NTHREADS, 0, stream>>>(
        coords, sp_tab, tm_tab, b1, g1, beta1, b2, g2, beta2, w3, b3,
        pw1, pw2, out);
}
